// AttentionBlock_34291018891756
// MI455X (gfx1250) — matrix operand order repaired and checked by reading, not compiled, not run on hardware
//
#include <hip/hip_runtime.h>
#include <hip/hip_bf16.h>

typedef __attribute__((ext_vector_type(16))) __bf16 v16bf;
typedef __attribute__((ext_vector_type(8)))  float  v8f;

#define WMMA_BF16(a,b,c) \
  __builtin_amdgcn_wmma_f32_16x16x32_bf16(false,(a),false,(b),(short)0,(c),false,false)

#if defined(__gfx1250__)
#define USE_ASYNC_LDS 1
#else
#define USE_ASYNC_LDS 0
#endif

__device__ inline v8f zero8(){ v8f z = {0.f,0.f,0.f,0.f,0.f,0.f,0.f,0.f}; return z; }

// A fragment: 16x32 bf16, A[M][K] row-major, leading dim ldk.
__device__ inline v16bf load_a_frag(const __bf16* base, int ldk, int kbase){
  const int lane  = threadIdx.x & 31;
  const int row   = lane & 15;
  const int khalf = (lane < 16) ? 0 : 16;
  const __bf16* p = base + (size_t)row * ldk + kbase;
  v16bf a;
#pragma unroll
  for (int r = 0; r < 8; ++r){
    int k = khalf + 2*r;
    a[2*r]   = p[k];
    a[2*r+1] = p[k+1];
  }
  return a;
}

// B fragment: 32x16 bf16 from Bt[N][K] (K-contiguous per column), leading dim ldk.
__device__ inline v16bf load_b_frag(const __bf16* B0, int ldk, int kbase, int col0, int maxcol){
  const int lane = threadIdx.x & 31;
  int col = col0 + (lane & 15);
  if (col > maxcol) col = maxcol;
  const __bf16* p = B0 + (size_t)col * ldk + kbase + ((lane < 16) ? 0 : 16);
  v16bf b;
#pragma unroll
  for (int e = 0; e < 16; ++e) b[e] = p[e];
  return b;
}

__device__ inline float hmax16(float v){
#pragma unroll
  for (int m = 1; m < 16; m <<= 1) v = fmaxf(v, __shfl_xor(v, m, 32));
  return v;
}
__device__ inline float hsum16(float v){
#pragma unroll
  for (int m = 1; m < 16; m <<= 1) v += __shfl_xor(v, m, 32);
  return v;
}

// 16B global -> LDS copy. On gfx1250 use the CDNA5 async-to-LDS path (ASYNCcnt);
// the destination VGPR carries the LDS byte offset (low 32 bits of generic LDS ptr).
__device__ inline void stage16(const __bf16* g, __bf16* l){
#if USE_ASYNC_LDS
  asm volatile("global_load_async_to_lds_b128 %0, %1, off"
               :: "v"((unsigned)(size_t)l), "v"(g)
               : "memory");
#else
  *(uint4*)l = *(const uint4*)g;
#endif
}

// ---------------- fp32 -> bf16 conversion ----------------
__global__ void f2bf_kernel(const float* __restrict__ in, __bf16* __restrict__ o, int n){
  int i = blockIdx.x * 256 + threadIdx.x;
  if (i < n) o[i] = (__bf16)in[i];
}

// ---------------- fused double GroupNorm ----------------
__device__ inline float block_reduce_256(float v, float* sm){
  int tid = threadIdx.x;
  sm[tid] = v; __syncthreads();
  for (int s = 128; s > 0; s >>= 1){
    if (tid < s) sm[tid] += sm[tid + s];
    __syncthreads();
  }
  float r = sm[0]; __syncthreads();
  return r;
}

__global__ void groupnorm2_kernel(const float* __restrict__ x,
                                  const float* __restrict__ g1, const float* __restrict__ b1,
                                  const float* __restrict__ g2, const float* __restrict__ b2,
                                  float* __restrict__ h32, __bf16* __restrict__ hn){
  const int grp = blockIdx.x, b = blockIdx.y, tid = threadIdx.x;
  const int CG = 16, L = 1024, C = 512, NE = CG * L;
  const float* xg = x + ((size_t)b * C + grp * CG) * L;
  __shared__ float sm[256];
  float s = 0.f, ss = 0.f;
  for (int i = tid; i < NE; i += 256){ float v = xg[i]; s += v; ss += v*v; }
  s  = block_reduce_256(s, sm);
  ss = block_reduce_256(ss, sm);
  float mu   = s / (float)NE;
  float rstd = rsqrtf(ss / (float)NE - mu*mu + 1e-5f);
  float s2 = 0.f, ss2 = 0.f;
  for (int i = tid; i < NE; i += 256){
    int c = grp*CG + (i >> 10);
    float hv = (xg[i]-mu)*rstd*g1[c] + b1[c];
    s2 += hv; ss2 += hv*hv;
  }
  s2  = block_reduce_256(s2, sm);
  ss2 = block_reduce_256(ss2, sm);
  float mu2   = s2 / (float)NE;
  float rstd2 = rsqrtf(ss2 / (float)NE - mu2*mu2 + 1e-5f);
  for (int i = tid; i < NE; i += 256){
    int c = grp*CG + (i >> 10);
    int l = i & 1023;
    float hv = (xg[i]-mu)*rstd*g1[c] + b1[c];
    size_t o = ((size_t)b * L + l) * C + c;
    h32[o] = hv;
    hn[o]  = (__bf16)((hv - mu2)*rstd2*g2[c] + b2[c]);
  }
}

// ---------------- LDS-staged GEMM ----------------
// Block = 4 waves = 64(M) x 64(N) tile. B tile (64 cols x 32 K, bf16 = 4KB) is staged
// through double-buffered LDS with async global->LDS copies; the 4 waves share it.
// out[b][n][m] = A[m][:]·B[b][n][:] + bias (+resid[b][n][m], act).
// omode 1: out32[(b*M+m)*N+n] = v + xres[(b*M+m)*N+n] (NCHW store with extra residual).
__global__ void __launch_bounds__(128)
gemm_bf16_kernel(const __bf16* __restrict__ A, const __bf16* __restrict__ B,
                 const float* __restrict__ bias, const float* __restrict__ resid,
                 const float* __restrict__ xres,
                 float* __restrict__ out32, __bf16* __restrict__ out16,
                 int M, int N, int K, int Nout, int act, int omode){
  const int n0 = blockIdx.x * 64;
  const int m0 = blockIdx.y * 64 + (threadIdx.x >> 5) * 16;
  const int b  = blockIdx.z;
  const int lane = threadIdx.x & 31;
  const int colL = lane & 15;
  const int rb   = (lane < 16) ? 0 : 8;
  const int tid  = threadIdx.x;
  const __bf16* Ab = A + (size_t)m0 * K;
  const __bf16* Bb = B + (size_t)b * N * K;
  __shared__ __attribute__((aligned(16))) __bf16 Bl[2][64*32];

  // stage one 64x32 B tile (2 x 16B chunks per thread)
  auto stage_tile = [&](int ks, __bf16* dst){
#pragma unroll
    for (int it = 0; it < 2; ++it){
      int c = tid + it*128;          // chunk id 0..255
      int col = c >> 2, seg = c & 3; // 4 x 16B per column
      int gcol = n0 + col; if (gcol > N-1) gcol = N-1;
      stage16(Bb + (size_t)gcol*K + ks + seg*8, dst + col*32 + seg*8);
    }
  };

  v8f acc[4];
#pragma unroll
  for (int t = 0; t < 4; ++t) acc[t] = zero8();

  stage_tile(0, Bl[0]);
  int cur = 0;
  for (int ks = 0; ks < K; ks += 32){
    if (ks + 32 < K){
      stage_tile(ks + 32, Bl[cur ^ 1]);
#if USE_ASYNC_LDS
      asm volatile("s_wait_asynccnt 0x2" ::: "memory");  // current tile landed
#endif
    } else {
#if USE_ASYNC_LDS
      asm volatile("s_wait_asynccnt 0x0" ::: "memory");
#endif
    }
    __syncthreads();
    v16bf a = load_a_frag(Ab, K, ks);
#pragma unroll
    for (int t = 0; t < 4; ++t)
      acc[t] = WMMA_BF16(a, load_b_frag(Bl[cur], 32, 0, t*16, 63), acc[t]);
    __syncthreads();                                     // all reads done before reuse
    cur ^= 1;
  }

#pragma unroll
  for (int t = 0; t < 4; ++t){
    int n = n0 + t*16 + colL;
    if (n >= N) continue;
    size_t idxSeq = ((size_t)b * Nout + n) * M;
#pragma unroll
    for (int r = 0; r < 8; ++r){
      int m = m0 + rb + r;
      float v = acc[t][r] + bias[m];
      if (act == 1) v = 0.5f * v * (1.0f + erff(v * 0.70710678118f));  // exact GELU
      size_t is = idxSeq + m;
      if (resid) v += resid[is];
      if (omode == 0){
        if (out32) out32[is] = v;
        if (out16) out16[is] = (__bf16)v;
      } else {
        size_t in2 = ((size_t)b * M + m) * N + n;
        out32[in2] = v + xres[in2];
      }
    }
  }
}

// ---------------- transposes ----------------
__global__ void transpose_v_kernel(const __bf16* __restrict__ qkv, __bf16* __restrict__ vT){
  size_t i = (size_t)blockIdx.x * 256 + threadIdx.x;
  int l = (int)(i & 1023);
  size_t j = i >> 10;
  int d = (int)(j & 63);
  int bh = (int)(j >> 6);
  int b = bh >> 3, h = bh & 7;
  vT[i] = qkv[((size_t)b*1024 + l)*1536 + 1024 + h*64 + d];
}

__global__ void transpose_v2_kernel(const __bf16* __restrict__ v2b, __bf16* __restrict__ v2t){
  size_t i = (size_t)blockIdx.x * 256 + threadIdx.x;
  int s = (int)(i % 96);
  size_t j = i / 96;
  int d = (int)(j & 63);
  int bh = (int)(j >> 6);
  int b = bh >> 3, h = bh & 7;
  v2t[i] = (s < 77) ? v2b[((size_t)b*80 + s)*512 + h*64 + d] : (__bf16)0.f;
}

// ---------------- flash self-attention ----------------
__global__ void __launch_bounds__(128)
self_attn_kernel(const __bf16* __restrict__ qkv, const __bf16* __restrict__ vT,
                 __bf16* __restrict__ aout){
  const int wid = threadIdx.x >> 5, lane = threadIdx.x & 31;
  const int tile = blockIdx.x * 4 + wid;
  const int bh = tile >> 6, l0 = (tile & 63) << 4;
  const int b = bh >> 3, h = bh & 7;
  const int colL = lane & 15, rb = (lane < 16) ? 0 : 8;
  __shared__ __bf16 Pb[4][16*32];
  __bf16* P = Pb[wid];
  const __bf16* Qb = qkv + ((size_t)(b*1024 + l0))*1536 + h*64;
  const __bf16* Kb = qkv + (size_t)b*1024*1536 + 512 + h*64;
  const __bf16* Vt = vT + (size_t)bh * 64 * 1024;
  v16bf q0 = load_a_frag(Qb, 1536, 0);
  v16bf q1 = load_a_frag(Qb, 1536, 32);
  v8f acc[4];
#pragma unroll
  for (int t = 0; t < 4; ++t) acc[t] = zero8();
  float mrow[8], lsum[8];
#pragma unroll
  for (int r = 0; r < 8; ++r){ mrow[r] = -1e30f; lsum[r] = 0.f; }

  for (int j = 0; j < 1024; j += 32){
    v8f s0 = zero8(), s1 = zero8();
    s0 = WMMA_BF16(q0, load_b_frag(Kb, 1536, 0,  j,      1023), s0);
    s0 = WMMA_BF16(q1, load_b_frag(Kb, 1536, 32, j,      1023), s0);
    s1 = WMMA_BF16(q0, load_b_frag(Kb, 1536, 0,  j + 16, 1023), s1);
    s1 = WMMA_BF16(q1, load_b_frag(Kb, 1536, 32, j + 16, 1023), s1);
    float scl[8];
#pragma unroll
    for (int r = 0; r < 8; ++r){
      float a0 = s0[r]*0.125f, a1 = s1[r]*0.125f;
      float mn = fmaxf(mrow[r], hmax16(fmaxf(a0, a1)));
      float esc = __expf(mrow[r] - mn);
      mrow[r] = mn;
      float p0 = __expf(a0 - mn), p1 = __expf(a1 - mn);
      lsum[r] = lsum[r]*esc + hsum16(p0 + p1);
      scl[r] = esc;
      int row = rb + r;
      P[row*32 + colL]      = (__bf16)p0;
      P[row*32 + 16 + colL] = (__bf16)p1;
    }
#pragma unroll
    for (int t = 0; t < 4; ++t)
#pragma unroll
      for (int r = 0; r < 8; ++r) acc[t][r] *= scl[r];
    asm volatile("s_wait_dscnt 0" ::: "memory");   // cross-lane LDS P visibility
    v16bf pa = load_a_frag(P, 32, 0);
#pragma unroll
    for (int t = 0; t < 4; ++t)
      acc[t] = WMMA_BF16(pa, load_b_frag(Vt, 1024, j, t*16, 63), acc[t]);
  }
#pragma unroll
  for (int t = 0; t < 4; ++t)
#pragma unroll
    for (int r = 0; r < 8; ++r){
      int row = l0 + rb + r;
      int d = t*16 + colL;
      aout[((size_t)b*1024 + row)*512 + h*64 + d] = (__bf16)(acc[t][r] / lsum[r]);
    }
}

// ---------------- cross-attention (S=77, padded to 80 keys / K=96) ----------------
__global__ void __launch_bounds__(128)
cross_attn_kernel(const __bf16* __restrict__ q2, const __bf16* __restrict__ k2,
                  const __bf16* __restrict__ v2t, __bf16* __restrict__ ca){
  const int wid = threadIdx.x >> 5, lane = threadIdx.x & 31;
  const int tile = blockIdx.x * 4 + wid;
  const int bh = tile >> 6, l0 = (tile & 63) << 4;
  const int b = bh >> 3, h = bh & 7;
  const int colL = lane & 15, rb = (lane < 16) ? 0 : 8;
  __shared__ __bf16 Pb[4][16*96];
  __bf16* P = Pb[wid];
  for (int i = lane; i < 256; i += 32)              // zero-pad columns 80..95
    P[(i >> 4)*96 + 80 + (i & 15)] = (__bf16)0.f;
  const __bf16* Qb = q2 + ((size_t)(b*1024 + l0))*512 + h*64;
  const __bf16* Kb = k2 + (size_t)b*80*512 + h*64;
  const __bf16* Vt = v2t + (size_t)bh * 64 * 96;
  v16bf q0 = load_a_frag(Qb, 512, 0);
  v16bf q1 = load_a_frag(Qb, 512, 32);
  v8f st[5];
#pragma unroll
  for (int kt = 0; kt < 5; ++kt){
    st[kt] = zero8();
    st[kt] = WMMA_BF16(q0, load_b_frag(Kb, 512, 0,  kt*16, 79), st[kt]);
    st[kt] = WMMA_BF16(q1, load_b_frag(Kb, 512, 32, kt*16, 79), st[kt]);
  }
  float lsum[8];
#pragma unroll
  for (int r = 0; r < 8; ++r){
    float mx = -1e30f;
#pragma unroll
    for (int kt = 0; kt < 5; ++kt){
      float v = st[kt][r]*0.125f;
      if (kt*16 + colL >= 77) v = -1e30f;          // mask padded keys
      st[kt][r] = v;
      mx = fmaxf(mx, v);
    }
    mx = hmax16(mx);
    float sum = 0.f;
    int row = rb + r;
#pragma unroll
    for (int kt = 0; kt < 5; ++kt){
      float p = __expf(st[kt][r] - mx);
      sum += p;
      P[row*96 + kt*16 + colL] = (__bf16)p;
    }
    lsum[r] = hsum16(sum);
  }
  asm volatile("s_wait_dscnt 0" ::: "memory");
  v8f acc[4];
#pragma unroll
  for (int t = 0; t < 4; ++t) acc[t] = zero8();
#pragma unroll
  for (int kk = 0; kk < 3; ++kk){
    v16bf pa = load_a_frag(P, 96, kk*32);
#pragma unroll
    for (int t = 0; t < 4; ++t)
      acc[t] = WMMA_BF16(pa, load_b_frag(Vt, 96, kk*32, t*16, 63), acc[t]);
  }
#pragma unroll
  for (int t = 0; t < 4; ++t)
#pragma unroll
    for (int r = 0; r < 8; ++r){
      int row = l0 + rb + r;
      int d = t*16 + colL;
      ca[((size_t)b*1024 + row)*512 + h*64 + d] = (__bf16)(acc[t][r] / lsum[r]);
    }
}

// ======================================================================
extern "C" void kernel_launch(void* const* d_in, const int* in_sizes, int n_in,
                              void* d_out, int out_size, void* d_ws, size_t ws_size,
                              hipStream_t stream){
  (void)in_sizes; (void)n_in; (void)out_size; (void)ws_size;
  const float* x      = (const float*)d_in[0];
  const float* ctx    = (const float*)d_in[1];
  const float* gn_g   = (const float*)d_in[2];
  const float* gn_b   = (const float*)d_in[3];
  const float* sa_g   = (const float*)d_in[4];
  const float* sa_b   = (const float*)d_in[5];
  const float* qkv_w  = (const float*)d_in[6];
  const float* qkv_b  = (const float*)d_in[7];
  const float* saw    = (const float*)d_in[8];
  const float* sab    = (const float*)d_in[9];
  const float* q_w    = (const float*)d_in[10];
  const float* q_b    = (const float*)d_in[11];
  const float* k_w    = (const float*)d_in[12];
  const float* k_b    = (const float*)d_in[13];
  const float* v_w    = (const float*)d_in[14];
  const float* v_b    = (const float*)d_in[15];
  const float* caw    = (const float*)d_in[16];
  const float* cab    = (const float*)d_in[17];
  const float* w1     = (const float*)d_in[18];
  const float* b1     = (const float*)d_in[19];
  const float* w2     = (const float*)d_in[20];
  const float* b2     = (const float*)d_in[21];
  float* out = (float*)d_out;

  char* ws = (char*)d_ws;
  size_t off = 0;
  auto bumpB = [&](size_t bytes)->void*{
    void* p = ws + off;
    off += (bytes + 255) & ~(size_t)255;
    return p;
  };
  __bf16* wqkv = (__bf16*)bumpB((size_t)1536*512*2);
  __bf16* wsa  = (__bf16*)bumpB((size_t)512*512*2);
  __bf16* wq   = (__bf16*)bumpB((size_t)512*512*2);
  __bf16* wk   = (__bf16*)bumpB((size_t)512*768*2);
  __bf16* wv   = (__bf16*)bumpB((size_t)512*768*2);
  __bf16* wca  = (__bf16*)bumpB((size_t)512*512*2);
  __bf16* w1b  = (__bf16*)bumpB((size_t)2048*512*2);
  __bf16* w2b  = (__bf16*)bumpB((size_t)512*2048*2);
  __bf16* ctxb = (__bf16*)bumpB((size_t)8*77*768*2);
  __bf16* hnb  = (__bf16*)bumpB((size_t)8*1024*512*2);
  float*  h32  = (float*) bumpB((size_t)8*1024*512*4);
  __bf16* qkvb = (__bf16*)bumpB((size_t)8*1024*1536*2);
  __bf16* vT   = (__bf16*)bumpB((size_t)8*8*64*1024*2);
  __bf16* aob  = (__bf16*)bumpB((size_t)8*1024*512*2);
  float*  h2f  = (float*) bumpB((size_t)8*1024*512*4);
  __bf16* h2b  = (__bf16*)bumpB((size_t)8*1024*512*2);
  __bf16* k2b  = (__bf16*)bumpB((size_t)8*80*512*2);
  __bf16* v2b  = (__bf16*)bumpB((size_t)8*80*512*2);
  __bf16* v2t  = (__bf16*)bumpB((size_t)8*8*64*96*2);
  __bf16* q2b  = (__bf16*)bumpB((size_t)8*1024*512*2);
  __bf16* cabf = (__bf16*)bumpB((size_t)8*1024*512*2);
  float*  h3f  = (float*) bumpB((size_t)8*1024*512*4);
  __bf16* h3b  = (__bf16*)bumpB((size_t)8*1024*512*2);
  __bf16* ff1  = (__bf16*)bumpB((size_t)8*1024*2048*2);

  auto cvt = [&](const float* in, __bf16* o, size_t n){
    f2bf_kernel<<<dim3((unsigned)((n + 255) / 256)), dim3(256), 0, stream>>>(in, o, (int)n);
  };
  cvt(qkv_w, wqkv, (size_t)1536*512);
  cvt(saw,   wsa,  (size_t)512*512);
  cvt(q_w,   wq,   (size_t)512*512);
  cvt(k_w,   wk,   (size_t)512*768);
  cvt(v_w,   wv,   (size_t)512*768);
  cvt(caw,   wca,  (size_t)512*512);
  cvt(w1,    w1b,  (size_t)2048*512);
  cvt(w2,    w2b,  (size_t)512*2048);
  cvt(ctx,   ctxb, (size_t)8*77*768);

  groupnorm2_kernel<<<dim3(32, 8), dim3(256), 0, stream>>>(x, gn_g, gn_b, sa_g, sa_b, h32, hnb);

  auto gemm = [&](const __bf16* A, const __bf16* Bm, const float* bias, const float* resid,
                  const float* xres, float* o32, __bf16* o16,
                  int M, int N, int K, int Nout, int act, int omode){
    gemm_bf16_kernel<<<dim3((N + 63) / 64, M / 64, 8), dim3(128), 0, stream>>>(
        A, Bm, bias, resid, xres, o32, o16, M, N, K, Nout, act, omode);
  };

  // qkv: [b][l][1536]
  gemm(wqkv, hnb, qkv_b, nullptr, nullptr, nullptr, qkvb, 1536, 1024, 512, 1024, 0, 0);
  transpose_v_kernel<<<dim3(16384), dim3(256), 0, stream>>>(qkvb, vT);
  self_attn_kernel<<<dim3(1024), dim3(128), 0, stream>>>(qkvb, vT, aob);
  // sa_proj + residual(h)
  gemm(wsa, aob, sab, h32, nullptr, h2f, h2b, 512, 1024, 512, 1024, 0, 0);
  // cross-attn K/V projections from context (N=77, rows padded to 80)
  gemm(wk, ctxb, k_b, nullptr, nullptr, nullptr, k2b, 512, 77, 768, 80, 0, 0);
  gemm(wv, ctxb, v_b, nullptr, nullptr, nullptr, v2b, 512, 77, 768, 80, 0, 0);
  transpose_v2_kernel<<<dim3(1536), dim3(256), 0, stream>>>(v2b, v2t);
  // Q projection
  gemm(wq, h2b, q_b, nullptr, nullptr, nullptr, q2b, 512, 1024, 512, 1024, 0, 0);
  cross_attn_kernel<<<dim3(1024), dim3(128), 0, stream>>>(q2b, k2b, v2t, cabf);
  // ca_proj + residual(h2)
  gemm(wca, cabf, cab, h2f, nullptr, h3f, h3b, 512, 1024, 512, 1024, 0, 0);
  // FFN1 + exact GELU
  gemm(w1b, h3b, b1, nullptr, nullptr, nullptr, ff1, 2048, 1024, 512, 1024, 1, 0);
  // FFN2 + bias + h3 residual + x residual, NCHW f32 output (omode=1)
  gemm(w2b, ff1, b2, h3f, x, out, nullptr, 512, 1024, 2048, 1024, 0, 1);
}
